// Attention_25692494364795
// MI455X (gfx1250) — matrix operand order repaired and checked by reading, not compiled, not run on hardware
//
#include <hip/hip_runtime.h>
#include <stdint.h>

#define SEQLEN 4096
#define MDIM   2048
#define NDIM3  6144
#define NHEADS 16
#define HDIM   128

typedef __attribute__((ext_vector_type(16))) __bf16 v16bf;
typedef __attribute__((ext_vector_type(8)))  float  v8f;

union Frag { v16bf v; unsigned u[8]; };

__device__ __forceinline__ unsigned short f2bf(float f) {
  union { float f; unsigned u; } c; c.f = f;
  unsigned u = c.u;
  unsigned r = (u + 0x7FFFu + ((u >> 16) & 1u)) >> 16;  // round-to-nearest-even
  return (unsigned short)r;
}
__device__ __forceinline__ unsigned pack2(float lo, float hi) {
  return (unsigned)f2bf(lo) | ((unsigned)f2bf(hi) << 16);
}

#define WMMA_BF16(a, b, c) \
  __builtin_amdgcn_wmma_f32_16x16x32_bf16(false, (a), false, (b), (short)0, (c), false, false)

// ---- CDNA5 async global->LDS copy (ASYNCcnt path), guarded ----------------
#if defined(__gfx1250__) && __has_builtin(__builtin_amdgcn_global_load_async_to_lds_b128)
#define HAVE_ASYNC_LDS 1
#else
#define HAVE_ASYNC_LDS 0
#endif

#if HAVE_ASYNC_LDS
typedef int v4i_vs __attribute__((vector_size(16)));
typedef __attribute__((address_space(1))) v4i_vs* as1_v4i;
typedef __attribute__((address_space(3))) v4i_vs* as3_v4i;

__device__ __forceinline__ void async_copy_b128(const void* g, void* lds) {
  __builtin_amdgcn_global_load_async_to_lds_b128(
      (as1_v4i)(void*)g, (as3_v4i)lds, 0, 0);
}
__device__ __forceinline__ void wait_async() {
#if __has_builtin(__builtin_amdgcn_s_wait_asynccnt)
  __builtin_amdgcn_s_wait_asynccnt(0);
#else
  asm volatile("s_wait_asynccnt 0" ::: "memory");
#endif
}
#endif

// ---------------------------------------------------------------------------
// Kernel 1: hidden = x @ w_attn + b_attn  (4096x2048 @ 2048x6144)
// Output scattered as bf16 into per-head Q (pre-scaled by 1/sqrt(128)), K, V.
// Block tile 128x128, K-step 32, 8 waves each computing a 32x64 sub-tile.
// ---------------------------------------------------------------------------
__global__ __launch_bounds__(256) void qkv_gemm_kernel(
    const float* __restrict__ x, const float* __restrict__ w,
    const float* __restrict__ bias,
    unsigned short* __restrict__ qb, unsigned short* __restrict__ kb,
    unsigned short* __restrict__ vb)
{
  __shared__ __align__(16) unsigned short As[128 * 34];  // [m][k], pitch 34 (bank pad)
  __shared__ __align__(16) unsigned short Bs[128 * 34];  // [n][k] (transposed), pitch 34

  const int tid  = threadIdx.x;
  const int wave = tid >> 5, lane = tid & 31;
  const int half = lane >> 4, l16 = lane & 15;
  const int waveM = wave >> 1, waveN = wave & 1;
  const int mBase = blockIdx.y * 128, nBase = blockIdx.x * 128;

  v8f acc[2][4];
#pragma unroll
  for (int i = 0; i < 2; ++i)
#pragma unroll
    for (int j = 0; j < 4; ++j) acc[i][j] = (v8f){0, 0, 0, 0, 0, 0, 0, 0};

  for (int kk = 0; kk < MDIM; kk += 32) {
    // Stage A tile (128x32 fp32 -> bf16), As[m][k]
#pragma unroll
    for (int i = 0; i < 4; ++i) {
      int idx = tid + i * 256;              // 1024 float4s: 128 rows x 8
      int row = idx >> 3, c4 = (idx & 7) * 4;
      float4 a = *(const float4*)(x + (size_t)(mBase + row) * MDIM + kk + c4);
      *(unsigned*)&As[row * 34 + c4]     = pack2(a.x, a.y);
      *(unsigned*)&As[row * 34 + c4 + 2] = pack2(a.z, a.w);
    }
    // Stage B tile (32x128 fp32 -> bf16 transposed), Bs[n][k].
    // Each item loads two consecutive k-rows and packs (k,k+1) pairs so the
    // LDS stores are dword-granular (no sub-word move traffic).
#pragma unroll
    for (int i = 0; i < 2; ++i) {
      int idx = tid + i * 256;              // 512 items: 16 k-pairs x 32 n4
      int kp2 = idx >> 5, c4 = (idx & 31) * 4;
      const float* bp = w + (size_t)(kk + 2 * kp2) * NDIM3 + nBase + c4;
      float4 x0 = *(const float4*)bp;
      float4 x1 = *(const float4*)(bp + NDIM3);
      *(unsigned*)&Bs[(c4 + 0) * 34 + 2 * kp2] = pack2(x0.x, x1.x);
      *(unsigned*)&Bs[(c4 + 1) * 34 + 2 * kp2] = pack2(x0.y, x1.y);
      *(unsigned*)&Bs[(c4 + 2) * 34 + 2 * kp2] = pack2(x0.z, x1.z);
      *(unsigned*)&Bs[(c4 + 3) * 34 + 2 * kp2] = pack2(x0.w, x1.w);
    }
    if (kk + 32 < MDIM) {
      __builtin_prefetch(x + (size_t)(mBase + (tid >> 1)) * MDIM + kk + 32, 0, 1);
      __builtin_prefetch(w + (size_t)(kk + 32 + (tid & 31)) * NDIM3 + nBase, 0, 1);
    }
    __syncthreads();

    // A 16x32 bf16 fragment: lanes 0-15 -> M=lane, VGPR i holds K pair
    // { (i<4 ? 2i : 2i+8) + half*8 }.
    Frag aF[2], bF[4];
#pragma unroll
    for (int mi = 0; mi < 2; ++mi) {
      int m = waveM * 32 + mi * 16 + l16;
#pragma unroll
      for (int i = 0; i < 8; ++i) {
        int k0 = ((i < 4) ? 2 * i : 2 * i + 8) + half * 8;
        aF[mi].u[i] = *(const unsigned*)&As[m * 34 + k0];
      }
    }
    // B 32x16 bf16 fragment: VGPR j holds K pair { (j<4 ? 2j : 2j+8) + half*8 }, N = lane&15.
#pragma unroll
    for (int ni = 0; ni < 4; ++ni) {
      int n = waveN * 64 + ni * 16 + l16;
#pragma unroll
      for (int j = 0; j < 8; ++j) {
        int k0 = ((j < 4) ? 2 * j : 2 * j + 8) + half * 8;
        bF[ni].u[j] = *(const unsigned*)&Bs[n * 34 + k0];
      }
    }
#pragma unroll
    for (int mi = 0; mi < 2; ++mi)
#pragma unroll
      for (int ni = 0; ni < 4; ++ni)
        acc[mi][ni] = WMMA_BF16(aF[mi].v, bF[ni].v, acc[mi][ni]);
    __syncthreads();
  }

  // Epilogue: bias, split into Q/K/V per-head [head][seq][dim] bf16 buffers.
#pragma unroll
  for (int mi = 0; mi < 2; ++mi) {
#pragma unroll
    for (int ni = 0; ni < 4; ++ni) {
      int col = nBase + waveN * 64 + ni * 16 + l16;
#pragma unroll
      for (int r = 0; r < 8; ++r) {
        int row = mBase + waveM * 32 + mi * 16 + half * 8 + r;
        float v = acc[mi][ni][r] + bias[col];
        unsigned short* dst; int c;
        if (col < MDIM)          { dst = qb; c = col;            v *= 0.08838834764831845f; }
        else if (col < 2 * MDIM) { dst = kb; c = col - MDIM; }
        else                     { dst = vb; c = col - 2 * MDIM; }
        int head = c >> 7, d = c & 127;
        dst[(size_t)head * SEQLEN * HDIM + (size_t)row * HDIM + d] = f2bf(v);
      }
    }
  }
}

// ---------------------------------------------------------------------------
// Kernel 2: causal flash attention per head. Block = (head, 128 query rows),
// 8 waves x 16 rows each. Keys in 32-wide chunks up to the causal limit;
// K tile staged via async global->LDS b128 (ASYNCcnt) when available.
// ---------------------------------------------------------------------------
__global__ __launch_bounds__(256) void flash_attn_kernel(
    const unsigned short* __restrict__ qb, const unsigned short* __restrict__ kb,
    const unsigned short* __restrict__ vb, unsigned short* __restrict__ zb)
{
  __shared__ __align__(16) unsigned short Kt[32 * 136];   // [key][hd], pitch 136 (16B rows)
  __shared__ __align__(16) unsigned short Vs[128 * 34];   // [hd][key], pitch 34
  __shared__ __align__(16) unsigned short Ps[8 * 16 * 34];// per-wave [row][key]

  const int tid  = threadIdx.x;
  const int wave = tid >> 5, lane = tid & 31;
  const int half = lane >> 4, l16 = lane & 15;
  const int qBase = blockIdx.x * 128;
  const int head  = blockIdx.y;
  const int qRow0 = qBase + wave * 16;

  const unsigned* qp = (const unsigned*)(qb + (size_t)head * SEQLEN * HDIM);
  const unsigned* kp = (const unsigned*)(kb + (size_t)head * SEQLEN * HDIM);
  const unsigned* vp = (const unsigned*)(vb + (size_t)head * SEQLEN * HDIM);

  // Q fragments (16 rows x 128 hd = 4 A-fragments) held in registers.
  Frag qF[4];
  {
    int row = qRow0 + l16;
#pragma unroll
    for (int c = 0; c < 4; ++c)
#pragma unroll
      for (int i = 0; i < 8; ++i) {
        int k0 = ((i < 4) ? 2 * i : 2 * i + 8) + half * 8 + c * 32;
        qF[c].u[i] = qp[(size_t)row * 64 + (k0 >> 1)];
      }
  }

  v8f o[8];
#pragma unroll
  for (int i = 0; i < 8; ++i) o[i] = (v8f){0, 0, 0, 0, 0, 0, 0, 0};
  float mrow[8], lrow[8];
#pragma unroll
  for (int r = 0; r < 8; ++r) { mrow[r] = -1e30f; lrow[r] = 0.0f; }

  const int nChunks = (qBase + 128) >> 5;   // causal limit for this block
  for (int kc = 0; kc < nChunks; ++kc) {
    const int kBase = kc * 32;
    __syncthreads();
    // Stage K chunk [32 keys][128 hd] bf16 -> Kt (straight copy)
#if HAVE_ASYNC_LDS
#pragma unroll
    for (int i = 0; i < 2; ++i) {
      int idx = tid + i * 256;              // 512 b128 chunks: 32 rows x 16
      int row = idx >> 4, c = idx & 15;
      async_copy_b128(kp + (size_t)(kBase + row) * 64 + c * 4,
                      (char*)Kt + row * 272 + c * 16);
    }
#else
#pragma unroll
    for (int i = 0; i < 8; ++i) {
      int idx = tid + i * 256;              // 2048 dwords
      int row = idx >> 6, dc = idx & 63;
      ((unsigned*)Kt)[row * 68 + dc] = kp[(size_t)(kBase + row) * 64 + dc];
    }
#endif
    // Stage V chunk transposed -> Vs[hd][key] (needs per-element transpose)
#pragma unroll
    for (int i = 0; i < 8; ++i) {
      int idx = tid + i * 256;
      int row = idx >> 6, dc = idx & 63;
      unsigned wv = vp[(size_t)(kBase + row) * 64 + dc];
      Vs[(2 * dc + 0) * 34 + row] = (unsigned short)(wv & 0xFFFFu);
      Vs[(2 * dc + 1) * 34 + row] = (unsigned short)(wv >> 16);
    }
    if (kc + 1 < nChunks)
      __builtin_prefetch(kp + (size_t)(kBase + 32 + (tid >> 3)) * 64, 0, 1);
#if HAVE_ASYNC_LDS
    wait_async();
#endif
    __syncthreads();

    if (kBase <= qRow0 + 15) {              // skip fully-masked chunks
      // S = Q K^T : two 16-key sub-tiles, contraction over hd in 4 steps
      v8f s[2];
      s[0] = (v8f){0, 0, 0, 0, 0, 0, 0, 0};
      s[1] = (v8f){0, 0, 0, 0, 0, 0, 0, 0};
#pragma unroll
      for (int nt = 0; nt < 2; ++nt) {
        int n = nt * 16 + l16;
#pragma unroll
        for (int c = 0; c < 4; ++c) {
          Frag bF;
#pragma unroll
          for (int j = 0; j < 8; ++j) {
            int k0 = c * 32 + ((j < 4) ? 2 * j : 2 * j + 8) + half * 8;
            bF.u[j] = *(const unsigned*)&Kt[n * 136 + k0];
          }
          s[nt] = WMMA_BF16(qF[c].v, bF.v, s[nt]);
        }
      }
      // Causal mask + online softmax update (rows live in 16-lane groups)
#pragma unroll
      for (int r = 0; r < 8; ++r) {
        int qr = qRow0 + half * 8 + r;
        float s0 = s[0][r], s1 = s[1][r];
        if (kBase + l16 > qr)      s0 = -1e30f;
        if (kBase + 16 + l16 > qr) s1 = -1e30f;
        float mc = fmaxf(s0, s1);
#pragma unroll
        for (int d = 8; d >= 1; d >>= 1) mc = fmaxf(mc, __shfl_xor(mc, d, 32));
        float mn = fmaxf(mrow[r], mc);
        float sc = __expf(mrow[r] - mn);
        mrow[r] = mn;
        float p0 = __expf(s0 - mn), p1 = __expf(s1 - mn);
        float ls = p0 + p1;
#pragma unroll
        for (int d = 8; d >= 1; d >>= 1) ls += __shfl_xor(ls, d, 32);
        lrow[r] = lrow[r] * sc + ls;
#pragma unroll
        for (int ht = 0; ht < 8; ++ht) o[ht][r] *= sc;
        int prow = half * 8 + r;
        Ps[(wave * 16 + prow) * 34 + l16]      = f2bf(p0);
        Ps[(wave * 16 + prow) * 34 + 16 + l16] = f2bf(p1);
      }
      // Re-read P in A-fragment layout (same wave: compiler handles DScnt)
      Frag pF;
#pragma unroll
      for (int i = 0; i < 8; ++i) {
        int k0 = ((i < 4) ? 2 * i : 2 * i + 8) + half * 8;
        pF.u[i] = *(const unsigned*)&Ps[(wave * 16 + l16) * 34 + k0];
      }
      // O += P @ V over 8 head-dim tiles
#pragma unroll
      for (int ht = 0; ht < 8; ++ht) {
        Frag vF;
#pragma unroll
        for (int j = 0; j < 8; ++j) {
          int k0 = ((j < 4) ? 2 * j : 2 * j + 8) + half * 8;
          vF.u[j] = *(const unsigned*)&Vs[(ht * 16 + l16) * 34 + k0];
        }
        o[ht] = WMMA_BF16(pF.v, vF.v, o[ht]);
      }
    }
  }

  // Normalize and write z as bf16 [seq][head*128 + d]
#pragma unroll
  for (int r = 0; r < 8; ++r) {
    float inv = 1.0f / lrow[r];
    int row = qRow0 + half * 8 + r;
#pragma unroll
    for (int ht = 0; ht < 8; ++ht) {
      int col = head * HDIM + ht * 16 + l16;
      zb[(size_t)row * MDIM + col] = f2bf(o[ht][r] * inv);
    }
  }
}

// ---------------------------------------------------------------------------
// Kernel 3: out = z @ w_proj + b_proj (A already bf16 -> async-stage to LDS)
// ---------------------------------------------------------------------------
__global__ __launch_bounds__(256) void proj_gemm_kernel(
    const unsigned short* __restrict__ zb, const float* __restrict__ w,
    const float* __restrict__ bias, float* __restrict__ out)
{
  __shared__ __align__(16) unsigned short As[128 * 40];  // [m][k], pitch 40 (16B rows)
  __shared__ __align__(16) unsigned short Bs[128 * 34];  // [n][k], pitch 34

  const int tid  = threadIdx.x;
  const int wave = tid >> 5, lane = tid & 31;
  const int half = lane >> 4, l16 = lane & 15;
  const int waveM = wave >> 1, waveN = wave & 1;
  const int mBase = blockIdx.y * 128, nBase = blockIdx.x * 128;

  v8f acc[2][4];
#pragma unroll
  for (int i = 0; i < 2; ++i)
#pragma unroll
    for (int j = 0; j < 4; ++j) acc[i][j] = (v8f){0, 0, 0, 0, 0, 0, 0, 0};

  for (int kk = 0; kk < MDIM; kk += 32) {
    // A tile already bf16: async copy when available, else dword copy
#if HAVE_ASYNC_LDS
#pragma unroll
    for (int i = 0; i < 2; ++i) {
      int idx = tid + i * 256;              // 512 b128 chunks: 128 rows x 4
      int row = idx >> 2, c = idx & 3;
      async_copy_b128(zb + (size_t)(mBase + row) * MDIM + kk + c * 32,
                      (char*)As + row * 80 + c * 16);
    }
#else
#pragma unroll
    for (int i = 0; i < 8; ++i) {
      int idx = tid + i * 256;              // 2048 dwords: 128 rows x 16
      int row = idx >> 4, dc = idx & 15;
      ((unsigned*)As)[row * 20 + dc] =
          ((const unsigned*)zb)[(size_t)(mBase + row) * (MDIM / 2) + (kk >> 1) + dc];
    }
#endif
    // B tile fp32 -> bf16 transposed, dword-granular stores
#pragma unroll
    for (int i = 0; i < 2; ++i) {
      int idx = tid + i * 256;              // 512 items: 16 k-pairs x 32 n4
      int kp2 = idx >> 5, c4 = (idx & 31) * 4;
      const float* bp = w + (size_t)(kk + 2 * kp2) * MDIM + nBase + c4;
      float4 x0 = *(const float4*)bp;
      float4 x1 = *(const float4*)(bp + MDIM);
      *(unsigned*)&Bs[(c4 + 0) * 34 + 2 * kp2] = pack2(x0.x, x1.x);
      *(unsigned*)&Bs[(c4 + 1) * 34 + 2 * kp2] = pack2(x0.y, x1.y);
      *(unsigned*)&Bs[(c4 + 2) * 34 + 2 * kp2] = pack2(x0.z, x1.z);
      *(unsigned*)&Bs[(c4 + 3) * 34 + 2 * kp2] = pack2(x0.w, x1.w);
    }
    if (kk + 32 < MDIM)
      __builtin_prefetch(w + (size_t)(kk + 32 + (tid & 31)) * MDIM + nBase, 0, 1);
#if HAVE_ASYNC_LDS
    wait_async();
#endif
    __syncthreads();

    Frag aF[2], bF[4];
#pragma unroll
    for (int mi = 0; mi < 2; ++mi) {
      int m = waveM * 32 + mi * 16 + l16;
#pragma unroll
      for (int i = 0; i < 8; ++i) {
        int k0 = ((i < 4) ? 2 * i : 2 * i + 8) + half * 8;
        aF[mi].u[i] = *(const unsigned*)&As[m * 40 + k0];
      }
    }
#pragma unroll
    for (int ni = 0; ni < 4; ++ni) {
      int n = waveN * 64 + ni * 16 + l16;
#pragma unroll
      for (int j = 0; j < 8; ++j) {
        int k0 = ((j < 4) ? 2 * j : 2 * j + 8) + half * 8;
        bF[ni].u[j] = *(const unsigned*)&Bs[n * 34 + k0];
      }
    }
#pragma unroll
    for (int mi = 0; mi < 2; ++mi)
#pragma unroll
      for (int ni = 0; ni < 4; ++ni)
        acc[mi][ni] = WMMA_BF16(aF[mi].v, bF[ni].v, acc[mi][ni]);
    __syncthreads();
  }

#pragma unroll
  for (int mi = 0; mi < 2; ++mi) {
#pragma unroll
    for (int ni = 0; ni < 4; ++ni) {
      int col = nBase + waveN * 64 + ni * 16 + l16;
#pragma unroll
      for (int r = 0; r < 8; ++r) {
        int row = mBase + waveM * 32 + mi * 16 + half * 8 + r;
        out[(size_t)row * MDIM + col] = acc[mi][ni][r] + bias[col];
      }
    }
  }
}

// ---------------------------------------------------------------------------
// Workspace layout (d_ws, needs 64 MB):
//   qb, kb, vb : [16][4096][128] bf16 (Q pre-scaled by 1/sqrt(128))
//   zb         : [4096][2048]    bf16
// ---------------------------------------------------------------------------
extern "C" void kernel_launch(void* const* d_in, const int* in_sizes, int n_in,
                              void* d_out, int out_size, void* d_ws, size_t ws_size,
                              hipStream_t stream) {
  (void)in_sizes; (void)n_in; (void)out_size; (void)ws_size;
  const float* x      = (const float*)d_in[0];
  const float* w_attn = (const float*)d_in[1];
  const float* b_attn = (const float*)d_in[2];
  const float* w_proj = (const float*)d_in[3];
  const float* b_proj = (const float*)d_in[4];
  float* out = (float*)d_out;

  const size_t HSZ = (size_t)NHEADS * SEQLEN * HDIM;
  unsigned short* qb = (unsigned short*)d_ws;
  unsigned short* kb = qb + HSZ;
  unsigned short* vb = kb + HSZ;
  unsigned short* zb = vb + HSZ;

  qkv_gemm_kernel<<<dim3(NDIM3 / 128, SEQLEN / 128), 256, 0, stream>>>(
      x, w_attn, b_attn, qb, kb, vb);
  flash_attn_kernel<<<dim3(SEQLEN / 128, NHEADS), 256, 0, stream>>>(
      qb, kb, vb, zb);
  proj_gemm_kernel<<<dim3(MDIM / 128, SEQLEN / 128), 256, 0, stream>>>(
      zb, w_proj, b_proj, out);
}
